// ShapeConv_78082505441804
// MI455X (gfx1250) — matrix operand order repaired and checked by reading, not compiled, not run on hardware
//
#include <hip/hip_runtime.h>
#include <math.h>

typedef _Float16 v16h __attribute__((ext_vector_type(16)));
typedef float    v8f  __attribute__((ext_vector_type(8)));

// Problem constants (fixed by the reference)
#define BATCH 32
#define CH    3
#define LEN   8192
#define OUT   128
#define KW    64
#define WPOS  (LEN - KW + 1)      // 8129
#define CT    1024                // time positions per block
#define XS    (CT + KW + 16)      // 1104: chunk + window tail + phantom-tile pad

// monotonic float -> uint encoding (order preserving); 0 encodes the minimum
__device__ __forceinline__ unsigned enc_f32(float f) {
    unsigned u = __float_as_uint(f);
    return (u & 0x80000000u) ? ~u : (u | 0x80000000u);
}
__device__ __forceinline__ float dec_f32(unsigned e) {
    unsigned u = (e & 0x80000000u) ? (e & 0x7fffffffu) : ~e;
    return __uint_as_float(u);
}

__global__ void shapeconv_init(unsigned* ws) {
    int i = blockIdx.x * blockDim.x + threadIdx.x;
    if (i < BATCH * OUT) ws[i] = 0u;
}

__global__ __launch_bounds__(256)
void shapeconv_main(const float* __restrict__ x,
                    const float* __restrict__ w,
                    unsigned* __restrict__ ws) {
    __shared__ _Float16 xh_s[CH * XS];
    __shared__ _Float16 xl_s[CH * XS];
    __shared__ float    sq_s[XS];
    __shared__ float    win_s[CT];
    __shared__ float    wn_s[OUT];

    const int b      = blockIdx.y;
    const int cstart = blockIdx.x * CT;
    const int clen   = min(CT, WPOS - cstart);   // >= 1 by grid construction
    const int tid    = threadIdx.x;

    // ---- Phase 1: stage hi/lo f16 x slice, window norms, weight norms ----
    for (int i = tid; i < XS; i += 256) {
        const int gi = cstart + i;
        float v0 = 0.f, v1 = 0.f, v2 = 0.f;
        if (gi < LEN) {
            const float* xb = x + (size_t)b * CH * LEN + gi;
            v0 = xb[0 * LEN];
            v1 = xb[1 * LEN];
            v2 = xb[2 * LEN];
        }
        const _Float16 h0 = (_Float16)v0, h1 = (_Float16)v1, h2 = (_Float16)v2;
        xh_s[0 * XS + i] = h0;
        xh_s[1 * XS + i] = h1;
        xh_s[2 * XS + i] = h2;
        xl_s[0 * XS + i] = (_Float16)(v0 - (float)h0);
        xl_s[1 * XS + i] = (_Float16)(v1 - (float)h1);
        xl_s[2 * XS + i] = (_Float16)(v2 - (float)h2);
        sq_s[i] = v0 * v0 + v1 * v1 + v2 * v2;
    }
    if (tid < OUT) {
        const float* wo = w + (size_t)tid * (CH * KW);
        float s = 0.f;
        #pragma unroll 8
        for (int i = 0; i < CH * KW; ++i) s += wo[i] * wo[i];
        wn_s[tid] = 0.5f * s;
    }
    __syncthreads();
    for (int t = tid; t < clen; t += 256) {
        float s = 0.f;
        #pragma unroll 8
        for (int j = 0; j < KW; ++j) s += sq_s[t + j];
        win_s[t] = 0.5f * s;
    }
    __syncthreads();

    // ---- Phase 2: split-precision f16 WMMA conv + fused norms + max ----
    const int wave  = tid >> 5;           // 8 waves: wave w owns filters [16w,16w+16)
    const int lane  = tid & 31;
    const int nrow  = lane & 15;          // N for B/C; M for A
    const int kh    = (lane >> 4) << 3;   // A-matrix K base per lane-half (0 / 8)
    const int kb    = (lane >> 4) << 3;   // B-matrix K base per lane-half (0 / 8)
    const int obase = wave << 4;

    // Register-resident B fragments (hi/lo) for the 6 k-chunks of 32
    v16h bh[6], bl[6];
    #pragma unroll
    for (int s = 0; s < 6; ++s) {
        const float* wp = w + (obase + nrow) * (CH * KW) + 32 * s + kb;
        #pragma unroll
        for (int i = 0; i < 16; ++i) {
            const float wv = wp[i + ((i >= 8) ? 8 : 0)];
            const _Float16 h = (_Float16)wv;
            bh[s][i] = h;
            bl[s][i] = (_Float16)(wv - (float)h);
        }
    }
    const float wn = wn_s[obase + nrow];
    float vmax = -INFINITY;

    const int nt = (clen + 15) >> 4;
    for (int ti = 0; ti < nt; ti += 2) {
        const int t0 = ti << 4;
        const int t1 = t0 + 16;           // may be phantom (zero-padded LDS)
        v8f acc0 = {};
        v8f acc1 = {};
        #pragma unroll
        for (int s = 0; s < 6; ++s) {
            const int c     = s >> 1;
            const int k0    = (s & 1) << 5;             // k base within channel
            const int base0 = c * XS + t0 + nrow + k0 + kh;
            const int base1 = base0 + 16;
            v16h a0h, a0l, a1h, a1l;
            #pragma unroll
            for (int i = 0; i < 16; ++i) {
                const int off = i + ((i >= 8) ? 8 : 0); // A K-layout within lane
                a0h[i] = xh_s[base0 + off];
                a0l[i] = xl_s[base0 + off];
                a1h[i] = xh_s[base1 + off];
                a1l[i] = xl_s[base1 + off];
            }
            acc0 = __builtin_amdgcn_wmma_f32_16x16x32_f16(
                       false, a0h, false, bh[s], (short)0, acc0, false, false);
            acc0 = __builtin_amdgcn_wmma_f32_16x16x32_f16(
                       false, a0h, false, bl[s], (short)0, acc0, false, false);
            acc0 = __builtin_amdgcn_wmma_f32_16x16x32_f16(
                       false, a0l, false, bh[s], (short)0, acc0, false, false);
            acc1 = __builtin_amdgcn_wmma_f32_16x16x32_f16(
                       false, a1h, false, bh[s], (short)0, acc1, false, false);
            acc1 = __builtin_amdgcn_wmma_f32_16x16x32_f16(
                       false, a1h, false, bl[s], (short)0, acc1, false, false);
            acc1 = __builtin_amdgcn_wmma_f32_16x16x32_f16(
                       false, a1l, false, bh[s], (short)0, acc1, false, false);
        }
        #pragma unroll
        for (int r = 0; r < 8; ++r) {
            const int m  = r + ((lane >> 4) << 3);      // C/D row this lane holds
            const int u0 = t0 + m;
            const int u1 = t1 + m;
            if (u0 < clen) vmax = fmaxf(vmax, acc0[r] - wn - win_s[u0]);
            if (u1 < clen) vmax = fmaxf(vmax, acc1[r] - wn - win_s[u1]);
        }
    }

    // combine the two lane-halves that hold the same filter column
    vmax = fmaxf(vmax, __shfl_xor(vmax, 16, 32));
    if (lane < 16)
        atomicMax(ws + b * OUT + obase + nrow, enc_f32(vmax));
}

__global__ void shapeconv_final(const unsigned* __restrict__ ws,
                                float* __restrict__ out) {
    int i = blockIdx.x * blockDim.x + threadIdx.x;
    if (i < BATCH * OUT) out[i] = -2.0f * dec_f32(ws[i]);
}

extern "C" void kernel_launch(void* const* d_in, const int* in_sizes, int n_in,
                              void* d_out, int out_size, void* d_ws, size_t ws_size,
                              hipStream_t stream) {
    const float* x = (const float*)d_in[0];
    const float* w = (const float*)d_in[1];
    float* out     = (float*)d_out;
    unsigned* ws   = (unsigned*)d_ws;

    shapeconv_init<<<(BATCH * OUT + 255) / 256, 256, 0, stream>>>(ws);

    const int nchunks = (WPOS + CT - 1) / CT;   // 8
    shapeconv_main<<<dim3(nchunks, BATCH), 256, 0, stream>>>(x, w, ws);

    shapeconv_final<<<(BATCH * OUT + 255) / 256, 256, 0, stream>>>(ws, out);
}
